// Mol_AFPLayer_18820546691272
// MI455X (gfx1250) — matrix operand order repaired and checked by reading, not compiled, not run on hardware
//
#include <hip/hip_runtime.h>
#include <math.h>

#define N_NODES 500000
#define G_SEG   25000
#define H_DIM   256
#define H3      (3*H_DIM)

typedef __attribute__((ext_vector_type(16))) __bf16 v16bf;
typedef __attribute__((ext_vector_type(8)))  float  v8f;

// ---- block-wide sum over 256 threads (wave32: shfl within wave, LDS across) ----
__device__ __forceinline__ float block_reduce_sum(float v, float* lds) {
    int tid = threadIdx.x, lane = tid & 31, wave = tid >> 5;
    #pragma unroll
    for (int o = 16; o > 0; o >>= 1) v += __shfl_xor(v, o, 32);
    __syncthreads();
    if (lane == 0) lds[wave] = v;
    __syncthreads();
    if (tid == 0) {
        float t = 0.f;
        #pragma unroll
        for (int i = 0; i < 8; ++i) t += lds[i];
        lds[0] = t;
    }
    __syncthreads();
    float r = lds[0];
    __syncthreads();
    return r;
}

// ---- 1) s = leaky_relu(super); ss[g] = s[g,:] . w_align[H:] ----
__global__ void k_prep_super(const float* __restrict__ sup,
                             const float* __restrict__ w_align,
                             float* __restrict__ s, float* __restrict__ ss) {
    __shared__ float lds[8];
    int g = blockIdx.x, tid = threadIdx.x;
    float x  = sup[(size_t)g*H_DIM + tid];
    float sv = x > 0.f ? x : 0.01f * x;           // jax leaky_relu slope 0.01
    s[(size_t)g*H_DIM + tid] = sv;
    float tot = block_reduce_sum(sv * w_align[H_DIM + tid], lds);
    if (tid == 0) ss[g] = tot;
}

// ---- 2) segment boundaries from sorted segment_ids ----
__global__ void k_seg_bounds(const int* __restrict__ seg, int* __restrict__ start) {
    int i = blockIdx.x*blockDim.x + threadIdx.x;
    if (i >= N_NODES) return;
    int si = seg[i];
    if (i == 0) { for (int g = 0; g <= si; ++g) start[g] = 0; }
    else { int sp = seg[i-1]; for (int g = sp+1; g <= si; ++g) start[g] = i; }
    if (i == N_NODES-1) { for (int g = si+1; g <= G_SEG; ++g) start[g] = N_NODES; }
}

// ---- 3) per-segment softmax + weighted node average  p[g] = sum attn_i*node[i,:]
// Phase A: one wave per row (no block syncs in the row loop).
__global__ void k_group(const float* __restrict__ node, const int* __restrict__ seg_start,
                        const float* __restrict__ w_align, const float* __restrict__ b_align,
                        const float* __restrict__ ss, float* __restrict__ score,
                        float* __restrict__ p, float* __restrict__ attn_out) {
    __shared__ float lds[8];
    __shared__ float lmax[8];
    int g = blockIdx.x, tid = threadIdx.x;
    int lane = tid & 31, wave = tid >> 5;
    int n0 = seg_start[g], n1 = seg_start[g+1];
    int cnt = n1 - n0;
    float b0  = b_align[0];
    float ssg = ss[g];

    // lane-local slice of wn = w_align[0:256]
    const float4* wn4 = (const float4*)w_align;
    float4 w0 = wn4[lane*2], w1 = wn4[lane*2 + 1];

    // Phase A: scores, one wave per row
    float mloc = -INFINITY;
    for (int r = wave; r < cnt; r += 8) {
        const float4* row4 = (const float4*)(node + (size_t)(n0 + r)*H_DIM);
        float4 x0 = row4[lane*2], x1 = row4[lane*2 + 1];
        float v = x0.x*w0.x + x0.y*w0.y + x0.z*w0.z + x0.w*w0.w
                + x1.x*w1.x + x1.y*w1.y + x1.z*w1.z + x1.w*w1.w;
        #pragma unroll
        for (int o = 16; o > 0; o >>= 1) v += __shfl_xor(v, o, 32);
        float sc = v + ssg + b0;
        sc = sc > 0.f ? sc : 0.01f * sc;
        if (lane == 0) score[n0 + r] = sc;
        mloc = fmaxf(mloc, sc);                  // identical across the wave
    }
    if (lane == 0) lmax[wave] = mloc;
    __syncthreads();
    float m = -INFINITY;
    #pragma unroll
    for (int i = 0; i < 8; ++i) m = fmaxf(m, lmax[i]);

    if (cnt == 0) { p[(size_t)g*H_DIM + tid] = 0.f; return; }

    // Phase B: denominator
    float part = 0.f;
    for (int i = tid; i < cnt; i += H_DIM) part += expf(score[n0 + i] - m);
    float denom = block_reduce_sum(part, lds);

    // Phase C: attn + weighted sum (node rows cache-hot from phase A)
    float pacc = 0.f;
    for (int r = 0; r < cnt; ++r) {
        int row = n0 + r;
        float a = expf(score[row] - m) / denom;
        if (tid == 0) attn_out[row] = a;
        pacc += a * node[(size_t)row*H_DIM + tid];
    }
    p[(size_t)g*H_DIM + tid] = pacc;
}

// ---- 4a) pack weight matrix B[256 x Nn] (f32, row-major) into bf16 WMMA
// B-fragment order: frag index (kt*2+hi)*Nn + n holds 16 bf16:
// elem 2v   = B[kt*32 + 16*hi + 2v    ][n]
// elem 2v+1 = B[kt*32 + 16*hi + 2v + 1][n]
__global__ void k_pack_b(const float* __restrict__ B, __bf16* __restrict__ Bp, int Nn) {
    int idx = blockIdx.x*blockDim.x + threadIdx.x;
    int total = (H_DIM/32) * 2 * Nn;
    if (idx >= total) return;
    int n  = idx % Nn;
    int t  = idx / Nn;              // kt*2 + hi
    int hi = t & 1, kt = t >> 1;
    int k0 = kt*32 + 16*hi;
    __bf16* dst = Bp + (size_t)idx * 16;
    #pragma unroll
    for (int v = 0; v < 8; ++v) {
        dst[2*v]   = (__bf16)B[(size_t)(k0 + 2*v    )*Nn + n];
        dst[2*v+1] = (__bf16)B[(size_t)(k0 + 2*v + 1)*Nn + n];
    }
}

// ---- 4b) bf16 WMMA GEMM: C[M x Nn] = act(A[M x 256] @ B + bias)
// One wave per 16x16 tile; A via float4 pairs, B via packed 32B fragments at
// compile-time strides (immediate offsets). Nn/ACT are template params.
template <int NN, int ACT>
__global__ void k_gemm_bf16(const float* __restrict__ A, const __bf16* __restrict__ Bp,
                            const float* __restrict__ bias, float* __restrict__ C,
                            int M) {
    const int K = H_DIM;
    int tid  = threadIdx.x;
    int lane = tid & 31, wave = tid >> 5;
    const int ntiles = NN >> 4;
    int mtiles = (M + 15) >> 4;
    int idx = blockIdx.x * (blockDim.x >> 5) + wave;
    if (idx >= ntiles * mtiles) return;            // whole wave exits together
    int tM = idx / ntiles, tN = idx % ntiles;
    int lid = lane & 15, hi = lane >> 4;
    int arow = tM*16 + lid;
    int crow = arow < M ? arow : (M - 1);          // clamp: OOB rows masked at store
    const float* Arow = A + (size_t)crow*K;
    int bcol = tN*16 + lid;
    // lane-fixed B base; per-step offset kt*2*NN fragments is a compile-time imm
    const v16bf* Bf = (const v16bf*)Bp + (size_t)hi*NN + bcol;
    v8f acc = {};
    #pragma unroll
    for (int kt = 0; kt < K/32; ++kt) {
        int k0 = kt*32;
        // A fragment: 16 consecutive floats at k0+16*hi
        float4 a0 = *(const float4*)(Arow + k0 + 16*hi);
        float4 a1 = *(const float4*)(Arow + k0 + 16*hi + 4);
        float4 a2 = *(const float4*)(Arow + k0 + 16*hi + 8);
        float4 a3 = *(const float4*)(Arow + k0 + 16*hi + 12);
        v16bf af;
        af[0]=(__bf16)a0.x; af[1]=(__bf16)a0.y; af[2]=(__bf16)a0.z; af[3]=(__bf16)a0.w;
        af[4]=(__bf16)a1.x; af[5]=(__bf16)a1.y; af[6]=(__bf16)a1.z; af[7]=(__bf16)a1.w;
        af[8]=(__bf16)a2.x; af[9]=(__bf16)a2.y; af[10]=(__bf16)a2.z; af[11]=(__bf16)a2.w;
        af[12]=(__bf16)a3.x; af[13]=(__bf16)a3.y; af[14]=(__bf16)a3.z; af[15]=(__bf16)a3.w;
        // B fragment: contiguous 32B, address = lane base + constant offset
        v16bf bfr = Bf[kt * 2 * NN];
        acc = __builtin_amdgcn_wmma_f32_16x16x32_bf16(
                  false, af, false, bfr, (short)0, acc, false, false);
    }
    // C/D layout: VGPR r -> (M = r + 8*hi, N = lid)
    float bz = bias[bcol];
    #pragma unroll
    for (int r = 0; r < 8; ++r) {
        int orow = tM*16 + r + 8*hi;
        if (orow < M) {
            float val = acc[r] + bz;
            if (ACT == 1) val = val > 0.f ? val : (expf(val) - 1.f);   // elu
            C[(size_t)orow*NN + bcol] = val;
        }
    }
}

// ---- 5) fused GRU cell + relu (float4 streaming) ----
__global__ void k_gru(const float* __restrict__ gi, const float* __restrict__ gh,
                      const float* __restrict__ s, float* __restrict__ out) {
    int i4 = blockIdx.x*blockDim.x + threadIdx.x;      // over G*H/4
    if (i4 >= G_SEG*(H_DIM/4)) return;
    int g  = i4 / (H_DIM/4);
    int j  = (i4 - g*(H_DIM/4)) * 4;
    size_t b = (size_t)g*H3;
    float4 ir = *(const float4*)(gi + b + j);
    float4 iz = *(const float4*)(gi + b + H_DIM + j);
    float4 in = *(const float4*)(gi + b + 2*H_DIM + j);
    float4 hr = *(const float4*)(gh + b + j);
    float4 hz = *(const float4*)(gh + b + H_DIM + j);
    float4 hn = *(const float4*)(gh + b + 2*H_DIM + j);
    float4 sv = *(const float4*)(s + (size_t)g*H_DIM + j);
    float4 o;
    #pragma unroll
    for (int c = 0; c < 4; ++c) {
        float irc = (&ir.x)[c], izc = (&iz.x)[c], inc = (&in.x)[c];
        float hrc = (&hr.x)[c], hzc = (&hz.x)[c], hnc = (&hn.x)[c];
        float r = 1.f/(1.f + expf(-(irc + hrc)));
        float z = 1.f/(1.f + expf(-(izc + hzc)));
        float n = tanhf(inc + r*hnc);
        float nh = (1.f - z)*n + z*(&sv.x)[c];
        (&o.x)[c] = nh > 0.f ? nh : 0.f;
    }
    *(float4*)(out + (size_t)g*H_DIM + j) = o;
}

extern "C" void kernel_launch(void* const* d_in, const int* in_sizes, int n_in,
                              void* d_out, int out_size, void* d_ws, size_t ws_size,
                              hipStream_t stream) {
    const float* node     = (const float*)d_in[0];
    const float* super_n  = (const float*)d_in[1];
    const int*   seg      = (const int*)  d_in[2];
    const float* w_align  = (const float*)d_in[3];
    const float* b_align  = (const float*)d_in[4];
    const float* w_attend = (const float*)d_in[5];
    const float* b_attend = (const float*)d_in[6];
    const float* w_ih     = (const float*)d_in[7];
    const float* w_hh     = (const float*)d_in[8];
    const float* b_ih     = (const float*)d_in[9];
    const float* b_hh     = (const float*)d_in[10];

    float* out_newh = (float*)d_out;                      // G*H
    float* out_attn = out_newh + (size_t)G_SEG*H_DIM;     // N

    float* ws_f = (float*)d_ws;
    size_t off = 0;
    float* s_buf   = ws_f + off; off += (size_t)G_SEG*H_DIM;
    float* ss_buf  = ws_f + off; off += G_SEG + 8;        // keep 32B alignment
    float* score   = ws_f + off; off += N_NODES;
    float* p_buf   = ws_f + off; off += (size_t)G_SEG*H_DIM;
    float* ctx_buf = ws_f + off; off += (size_t)G_SEG*H_DIM;
    float* gi_buf  = ws_f + off; off += (size_t)G_SEG*H3;
    float* gh_buf  = ws_f + off; off += (size_t)G_SEG*H3;
    int*   seg_start = (int*)(ws_f + off); off += G_SEG + 8;
    __bf16* bp_att = (__bf16*)(ws_f + off); off += (8*2*H_DIM*16)/2;  // 65536 bf16
    __bf16* bp_ih  = (__bf16*)(ws_f + off); off += (8*2*H3*16)/2;     // 196608 bf16
    __bf16* bp_hh  = (__bf16*)(ws_f + off); off += (8*2*H3*16)/2;

    k_prep_super<<<G_SEG, H_DIM, 0, stream>>>(super_n, w_align, s_buf, ss_buf);
    k_seg_bounds<<<(N_NODES+255)/256, 256, 0, stream>>>(seg, seg_start);

    // pack the three weight matrices into bf16 WMMA fragment order (once, tiny)
    k_pack_b<<<(8*2*H_DIM + 255)/256, 256, 0, stream>>>(w_attend, bp_att, H_DIM);
    k_pack_b<<<(8*2*H3   + 255)/256, 256, 0, stream>>>(w_ih,     bp_ih,  H3);
    k_pack_b<<<(8*2*H3   + 255)/256, 256, 0, stream>>>(w_hh,     bp_hh,  H3);

    k_group<<<G_SEG, H_DIM, 0, stream>>>(node, seg_start, w_align, b_align,
                                         ss_buf, score, p_buf, out_attn);
    {   // ctx = elu(p @ w_attend + b_attend)
        int tiles = ((G_SEG+15)/16) * (H_DIM/16);
        k_gemm_bf16<H_DIM, 1><<<(tiles+7)/8, 256, 0, stream>>>(p_buf, bp_att, b_attend,
                                                               ctx_buf, G_SEG);
    }
    {   // gi = ctx @ w_ih + b_ih ; gh = s @ w_hh + b_hh
        int tiles = ((G_SEG+15)/16) * (H3/16);
        k_gemm_bf16<H3, 0><<<(tiles+7)/8, 256, 0, stream>>>(ctx_buf, bp_ih, b_ih,
                                                            gi_buf, G_SEG);
        k_gemm_bf16<H3, 0><<<(tiles+7)/8, 256, 0, stream>>>(s_buf, bp_hh, b_hh,
                                                            gh_buf, G_SEG);
    }
    k_gru<<<(G_SEG*(H_DIM/4)+255)/256, 256, 0, stream>>>(gi_buf, gh_buf, s_buf, out_newh);
}
